// Net_gaussian_correction_34591666602133
// MI455X (gfx1250) — matrix operand order repaired and checked by reading, not compiled, not run on hardware
//
#include <hip/hip_runtime.h>
#include <hip/hip_bf16.h>

#define N_NODES 100000
#define N_EDGES 1600000
#define F_INF   27
#define EMB     64
#define HDIM    128
#define LAYERS  4
#define NGRAPH  512

typedef __attribute__((ext_vector_type(16))) _Float16 v16h;
typedef __attribute__((ext_vector_type(8)))  float    v8f;

__device__ __forceinline__ void atomAddF(float* p, float v) {
    __hip_atomic_fetch_add(p, v, __ATOMIC_RELAXED, __HIP_MEMORY_SCOPE_AGENT);
}

// ---------------------------------------------------------------------------
// Weight prep: f32 -> f16, transpose ggc weights so B-fragments are contiguous.
// Wt[l][n][k] = ggc[l][k][n];  wih16/whh16 are straight copies (already [Ncols,K]).
// ---------------------------------------------------------------------------
__global__ void prep_weights_kernel(const float* __restrict__ ggc,
                                    const float* __restrict__ w_ih,
                                    const float* __restrict__ w_hh,
                                    _Float16* __restrict__ Wt,
                                    _Float16* __restrict__ wih16,
                                    _Float16* __restrict__ whh16) {
    const int WSZ = HDIM * HDIM;       // 16384
    const int GSZ = 3 * HDIM * HDIM;   // 49152
    int i = blockIdx.x * blockDim.x + threadIdx.x;
    if (i < LAYERS * WSZ) {
        int l = i / WSZ, r = i - l * WSZ;
        int n = r >> 7, k = r & 127;
        Wt[i] = (_Float16)ggc[l * WSZ + k * HDIM + n];
    } else if (i < LAYERS * WSZ + GSZ) {
        int j = i - LAYERS * WSZ;
        wih16[j] = (_Float16)w_ih[j];
    } else if (i < LAYERS * WSZ + 2 * GSZ) {
        int j = i - LAYERS * WSZ - GSZ;
        whh16[j] = (_Float16)w_hh[j];
    }
}

// ---------------------------------------------------------------------------
// x1 = sigmoid(x @ lin0_w) ; h = pad(x1, 128)
// ---------------------------------------------------------------------------
__global__ void lin0_kernel(const float* __restrict__ x,
                            const float* __restrict__ w0,
                            float* __restrict__ x1_out,
                            float* __restrict__ h) {
    __shared__ float w[F_INF * EMB];
    for (int i = threadIdx.x; i < F_INF * EMB; i += blockDim.x) w[i] = w0[i];
    __syncthreads();
    int n = blockIdx.x * blockDim.x + threadIdx.x;
    if (n >= N_NODES) return;
    float xv[F_INF];
#pragma unroll
    for (int k = 0; k < F_INF; k++) xv[k] = x[(size_t)n * F_INF + k];
    float* hp = h + (size_t)n * HDIM;
    float* xo = x1_out + (size_t)n * EMB;
    for (int o = 0; o < EMB; o++) {
        float s = 0.0f;
#pragma unroll
        for (int k = 0; k < F_INF; k++) s += xv[k] * w[k * EMB + o];
        s = 1.0f / (1.0f + expf(-s));
        xo[o] = s;
        hp[o] = s;
    }
#pragma unroll
    for (int o = EMB; o < HDIM; o++) hp[o] = 0.0f;
}

// ---------------------------------------------------------------------------
// WMMA GEMM: C[rows, cols@ldc] (+)= cvt_f16(A[rows,128]) @ Bt^T (+ bias)
//   Bt is f16 [ncols, 128] row-major (i.e. B transposed), so each lane's
//   B-fragment is one contiguous 32B load.
//   One wave -> 16 rows x 64 cols (4 accumulators, A-fragment reused 4x).
// ---------------------------------------------------------------------------
__global__ void gemm_wmma_kernel(const float* __restrict__ A,
                                 const _Float16* __restrict__ Bt,
                                 float* __restrict__ C, int ldc,
                                 const float* __restrict__ bias,
                                 int cinit, int nRowTiles) {
    int wave = threadIdx.x >> 5;
    int lane = threadIdx.x & 31;
    int rt = blockIdx.x * 8 + wave;
    if (rt >= nRowTiles) return;              // wave-uniform: EXEC stays all-1s
    int ct0 = blockIdx.y * 64;
    int hi = lane >> 4;                       // lane half
    int lx = lane & 15;
    const float* ap = A + (size_t)(rt * 16 + lx) * HDIM + (hi ? 16 : 0);
    int rowC0 = rt * 16 + (hi ? 8 : 0);

    v8f acc[4];
#pragma unroll
    for (int t = 0; t < 4; t++) {
        if (cinit) {
            int col = ct0 + t * 16 + lx;
#pragma unroll
            for (int j = 0; j < 8; j++)
                acc[t][j] = C[(size_t)(rowC0 + j) * ldc + col];
        } else {
#pragma unroll
            for (int j = 0; j < 8; j++) acc[t][j] = 0.0f;
        }
    }

#pragma unroll
    for (int k0 = 0; k0 < HDIM; k0 += 32) {
        // A fragment (16x32 f16): lane<16 holds K {k0..k0+15},
        // lane>=16 holds K {k0+16..k0+31} of row (rt*16+lx).
        const float* app = ap + k0;
        float4 f0 = *(const float4*)(app);
        float4 f1 = *(const float4*)(app + 4);
        float4 f2 = *(const float4*)(app + 8);
        float4 f3 = *(const float4*)(app + 12);
        v16h a;
        a[0]  = (_Float16)f0.x; a[1]  = (_Float16)f0.y; a[2]  = (_Float16)f0.z; a[3]  = (_Float16)f0.w;
        a[4]  = (_Float16)f1.x; a[5]  = (_Float16)f1.y; a[6]  = (_Float16)f1.z; a[7]  = (_Float16)f1.w;
        a[8]  = (_Float16)f2.x; a[9]  = (_Float16)f2.y; a[10] = (_Float16)f2.z; a[11] = (_Float16)f2.w;
        a[12] = (_Float16)f3.x; a[13] = (_Float16)f3.y; a[14] = (_Float16)f3.z; a[15] = (_Float16)f3.w;
        int kb = k0 + (hi ? 16 : 0);
#pragma unroll
        for (int t = 0; t < 4; t++) {
            // B fragment (32x16 f16): lane half holds 16 contiguous K of col.
            int col = ct0 + t * 16 + lx;
            v16h b = *(const v16h*)(Bt + (size_t)col * HDIM + kb);
            acc[t] = __builtin_amdgcn_wmma_f32_16x16x32_f16(
                false, a, false, b, (short)0, acc[t], false, false);
        }
    }

#pragma unroll
    for (int t = 0; t < 4; t++) {
        int col = ct0 + t * 16 + lx;
        float bv = bias ? bias[col] : 0.0f;
#pragma unroll
        for (int j = 0; j < 8; j++)
            C[(size_t)(rowC0 + j) * ldc + col] = acc[t][j] + bv;
    }
}

// ---------------------------------------------------------------------------
// agg[dst[e]] += m[src[e]]  (one 32-lane group per edge, float4 per lane)
// ---------------------------------------------------------------------------
__global__ void aggregate_kernel(const float* __restrict__ m,
                                 const int* __restrict__ src,
                                 const int* __restrict__ dst,
                                 float* __restrict__ agg) {
    long long g = ((long long)blockIdx.x * blockDim.x + threadIdx.x) >> 5;
    if (g >= N_EDGES) return;
    int lane = threadIdx.x & 31;
    int s = src[g], d = dst[g];
    float4 v = *(const float4*)(m + (size_t)s * HDIM + lane * 4);
    float* dp = agg + (size_t)d * HDIM + lane * 4;
    atomAddF(dp + 0, v.x);
    atomAddF(dp + 1, v.y);
    atomAddF(dp + 2, v.z);
    atomAddF(dp + 3, v.w);
}

// ---------------------------------------------------------------------------
// GRU pointwise: P[:, :128]=r_sum, P[:,128:256]=z_sum, P[:,256:]=i_n(+b_ih_n)
// hn = h @ Whh_n^T + b_hh_n
// ---------------------------------------------------------------------------
__global__ void gru_update_kernel(const float* __restrict__ P,
                                  const float* __restrict__ hn,
                                  float* __restrict__ h) {
    size_t idx = (size_t)blockIdx.x * blockDim.x + threadIdx.x;
    if (idx >= (size_t)N_NODES * HDIM) return;
    size_t n = idx >> 7;
    int j = (int)(idx & 127);
    const float* p = P + n * (size_t)(3 * HDIM);
    float r  = 1.0f / (1.0f + expf(-p[j]));
    float z  = 1.0f / (1.0f + expf(-p[HDIM + j]));
    float nn = tanhf(p[2 * HDIM + j] + r * hn[idx]);
    h[idx] = (1.0f - z) * nn + z * h[idx];
}

__global__ void zero_kernel(float* __restrict__ p, size_t n) {
    size_t i = (size_t)blockIdx.x * blockDim.x + threadIdx.x;
    if (i < n) p[i] = 0.0f;
}

// ---------------------------------------------------------------------------
// Readout: mu / sigma per node + per-graph atomic sums
// ---------------------------------------------------------------------------
__global__ void readout_kernel(const float* __restrict__ h,
                               const float* __restrict__ l1w, const float* __restrict__ l1b,
                               const float* __restrict__ l2w, const float* __restrict__ l2b,
                               const int* __restrict__ batch,
                               float* __restrict__ out_mu, float* __restrict__ out_sigma,
                               float* __restrict__ mu_s, float* __restrict__ sig_s,
                               float* __restrict__ mu_all, float* __restrict__ sig_all) {
    __shared__ float w1[HDIM];
    __shared__ float w2[HDIM];
    int tid = threadIdx.x;
    if (tid < HDIM) w1[tid] = l1w[tid];
    else if (tid < 2 * HDIM) w2[tid - HDIM] = l2w[tid - HDIM];
    __syncthreads();
    int n = blockIdx.x * blockDim.x + tid;
    if (n >= N_NODES) return;
    const float4* hp = (const float4*)(h + (size_t)n * HDIM);
    float mu = 0.0f, sg = 0.0f;
#pragma unroll 8
    for (int j = 0; j < 32; j++) {
        float4 hv = hp[j];
        float a = fmaxf(hv.x, 0.0f), b = fmaxf(hv.y, 0.0f);
        float c = fmaxf(hv.z, 0.0f), d = fmaxf(hv.w, 0.0f);
        int k = j * 4;
        mu += a * w1[k] + b * w1[k + 1] + c * w1[k + 2] + d * w1[k + 3];
        sg += a * w2[k] + b * w2[k + 1] + c * w2[k + 2] + d * w2[k + 3];
    }
    mu += l1b[0];
    sg += l2b[0];
    float sigma = (sg > 20.0f) ? sg : log1pf(expf(sg));   // softplus
    out_mu[n] = mu;
    out_sigma[n] = sigma;
    mu_s[n] = mu;
    sig_s[n] = sigma;
    int b = batch[n];
    atomAddF(mu_all + b, mu);
    atomAddF(sig_all + b, sigma);
}

__global__ void correction_kernel(const float* __restrict__ mu_s,
                                  const float* __restrict__ sig_s,
                                  const float* __restrict__ mu_all,
                                  const float* __restrict__ sig_all,
                                  const int* __restrict__ batch,
                                  float* __restrict__ out) {
    int n = blockIdx.x * blockDim.x + threadIdx.x;
    if (n >= N_NODES) return;
    int b = batch[n];
    out[n] = mu_s[n] - mu_all[b] * (sig_s[n] / sig_all[b]);
}

// ---------------------------------------------------------------------------
extern "C" void kernel_launch(void* const* d_in, const int* in_sizes, int n_in,
                              void* d_out, int out_size, void* d_ws, size_t ws_size,
                              hipStream_t stream) {
    (void)in_sizes; (void)n_in; (void)out_size; (void)ws_size;
    const float* x      = (const float*)d_in[0];
    const int*   ei     = (const int*)d_in[1];
    const int*   batch  = (const int*)d_in[2];
    const float* lin0_w = (const float*)d_in[3];
    const float* ggc    = (const float*)d_in[4];
    const float* w_ih   = (const float*)d_in[5];
    const float* w_hh   = (const float*)d_in[6];
    const float* b_ih   = (const float*)d_in[7];
    const float* b_hh   = (const float*)d_in[8];
    const float* lin1_w = (const float*)d_in[9];
    const float* lin1_b = (const float*)d_in[10];
    const float* lin2_w = (const float*)d_in[11];
    const float* lin2_b = (const float*)d_in[12];

    float* out       = (float*)d_out;
    float* out_muc   = out;
    float* out_x1    = out + N_NODES;
    float* out_sigma = out + N_NODES + (size_t)N_NODES * EMB;
    float* out_mu    = out_sigma + N_NODES;

    // workspace layout (f32 first, then f16)
    float* w = (float*)d_ws;
    float* h    = w;  w += (size_t)N_NODES * HDIM;
    float* agg  = w;  w += (size_t)N_NODES * HDIM;
    float* P    = w;  w += (size_t)N_NODES * 3 * HDIM;
    float* hn   = w;  w += (size_t)N_NODES * HDIM;   // doubles as m
    float* mu_s = w;  w += N_NODES;
    float* sg_s = w;  w += N_NODES;
    float* mu_a = w;  w += NGRAPH;
    float* sg_a = w;  w += NGRAPH;
    _Float16* Wt    = (_Float16*)w;
    _Float16* wih16 = Wt + (size_t)LAYERS * HDIM * HDIM;
    _Float16* whh16 = wih16 + (size_t)3 * HDIM * HDIM;

    const int RT = N_NODES / 16;   // 6250 row tiles (exact)
    const dim3 gB(256);

    { // weight prep
        int tot = LAYERS * HDIM * HDIM + 2 * 3 * HDIM * HDIM;
        prep_weights_kernel<<<(tot + 255) / 256, gB, 0, stream>>>(ggc, w_ih, w_hh, Wt, wih16, whh16);
    }
    lin0_kernel<<<(N_NODES + 255) / 256, gB, 0, stream>>>(x, lin0_w, out_x1, h);

    const size_t NH = (size_t)N_NODES * HDIM;
    for (int l = 0; l < LAYERS; l++) {
        // m = h @ W_l            -> hn buffer
        gemm_wmma_kernel<<<dim3((RT + 7) / 8, 2), gB, 0, stream>>>(
            h, Wt + (size_t)l * HDIM * HDIM, hn, HDIM, nullptr, 0, RT);
        // agg = segment_sum(m[src], dst)
        zero_kernel<<<(unsigned)((NH + 255) / 256), gB, 0, stream>>>(agg, NH);
        aggregate_kernel<<<(unsigned)(((size_t)N_EDGES * 32 + 255) / 256), gB, 0, stream>>>(
            hn, ei, ei + N_EDGES, agg);
        // P = agg @ Wih^T + b_ih                     (all 384 cols)
        gemm_wmma_kernel<<<dim3((RT + 7) / 8, 6), gB, 0, stream>>>(
            agg, wih16, P, 3 * HDIM, b_ih, 0, RT);
        // P[:, :256] += h @ Whh_rz^T + b_hh_rz       (C-init accumulate)
        gemm_wmma_kernel<<<dim3((RT + 7) / 8, 4), gB, 0, stream>>>(
            h, whh16, P, 3 * HDIM, b_hh, 1, RT);
        // hn = h @ Whh_n^T + b_hh_n
        gemm_wmma_kernel<<<dim3((RT + 7) / 8, 2), gB, 0, stream>>>(
            h, whh16 + (size_t)2 * HDIM * HDIM, hn, HDIM, b_hh + 2 * HDIM, 0, RT);
        // GRU pointwise update (in place on h)
        gru_update_kernel<<<(unsigned)((NH + 255) / 256), gB, 0, stream>>>(P, hn, h);
    }

    zero_kernel<<<(2 * NGRAPH + 255) / 256, gB, 0, stream>>>(mu_a, 2 * NGRAPH);
    readout_kernel<<<(N_NODES + 255) / 256, gB, 0, stream>>>(
        h, lin1_w, lin1_b, lin2_w, lin2_b, batch,
        out_mu, out_sigma, mu_s, sg_s, mu_a, sg_a);
    correction_kernel<<<(N_NODES + 255) / 256, gB, 0, stream>>>(
        mu_s, sg_s, mu_a, sg_a, batch, out_muc);
}
